// PathAttention_75333726372356
// MI455X (gfx1250) — matrix operand order repaired and checked by reading, not compiled, not run on hardware
//
#include <hip/hip_runtime.h>
#include <hip/hip_bf16.h>

typedef __bf16 bf16;
typedef __attribute__((ext_vector_type(16))) __bf16 v16bf;
typedef __attribute__((ext_vector_type(8)))  __bf16 v8bf;
typedef __attribute__((ext_vector_type(8)))  float  v8f;

constexpr int BB    = 8;
constexpr int NSEQ  = 1024;
constexpr int HH    = 768;
constexpr int NHEAD = 12;
constexpr int ASZ   = 64;
constexpr int MOCC  = 256;
constexpr float QSCALE    = 0.125f;   // 64^-0.5
constexpr float NEG_SLOPE = 0.02f;

enum { ACT_NONE = 0, ACT_TANH = 1 };
enum { SM_F32 = 0, SM_BF16 = 1, SM_HEADS = 2, SM_HEADS_T = 3 };

// ---------------------------------------------------------------------------
// Generic bf16 WMMA GEMM:  C = act((A @ Bw^T + bias) * alpha)
// A: [M,K] bf16 row-major (lda), Bw: [Nc,K] bf16 row-major (ldb) -> B = Bw^T
// Block = 4 waves; wave tile = 16 rows x 64 cols. M%64==0, Nc%64==0, K%32==0.
// ---------------------------------------------------------------------------
template<int ACT, int SM, int ACCUM>
__global__ __launch_bounds__(128)
void gemm_bf16_wmma(const bf16* __restrict__ A, int lda, long sA,
                    const bf16* __restrict__ Bw, int ldb, long sB,
                    void* __restrict__ C, int ldc, long sC,
                    const float* __restrict__ bias, float alpha, int K)
{
    const int tid  = threadIdx.x;
    const int wv   = tid >> 5;
    const int lane = tid & 31;
    const int hl   = lane >> 4;
    const int ln   = lane & 15;
    const int rowBase = blockIdx.y * 64 + wv * 16;
    const int colBase = blockIdx.x * 64;
    const int bz = blockIdx.z;
    A  += (long)bz * sA;
    Bw += (long)bz * sB;

    v8f acc[4] = {};
    const bf16* arow = A + (long)(rowBase + ln) * lda;

    for (int kk = 0; kk < K; kk += 32) {
        union { v16bf v; v8bf h[2]; } au;
        au.h[0] = *(const v8bf*)(arow + kk + hl * 16);
        au.h[1] = *(const v8bf*)(arow + kk + hl * 16 + 8);
#pragma unroll
        for (int s = 0; s < 4; ++s) {
            const bf16* brow = Bw + (long)(colBase + s * 16 + ln) * ldb + kk + hl * 16;
            v16bf bfrag = *(const v16bf*)brow;
            acc[s] = __builtin_amdgcn_wmma_f32_16x16x32_bf16(
                false, au.v, false, bfrag, (short)0, acc[s], false, false);
        }
    }

#pragma unroll
    for (int s = 0; s < 4; ++s) {
        const int col  = colBase + s * 16 + ln;
        const float bv = bias ? bias[col] : 0.0f;
#pragma unroll
        for (int r = 0; r < 8; ++r) {
            const int row = rowBase + r + hl * 8;
            float v = (acc[s][r] + bv) * alpha;
            if (ACT == ACT_TANH) v = tanhf(v);
            if (SM == SM_F32) {
                float* p = (float*)C + (long)bz * sC + (long)row * ldc + col;
                if (ACCUM) *p += v; else *p = v;
            } else if (SM == SM_BF16) {
                ((bf16*)C)[(long)bz * sC + (long)row * ldc + col] = (bf16)v;
            } else if (SM == SM_HEADS) {      // [B, NH, N, AS]
                const int b = row >> 10, n = row & (NSEQ - 1);
                const int h = col >> 6,  d = col & (ASZ - 1);
                ((bf16*)C)[(((long)(b * NHEAD + h)) * NSEQ + n) * ASZ + d] = (bf16)v;
            } else {                          // [B, NH, AS, N]  (V transposed)
                const int b = row >> 10, n = row & (NSEQ - 1);
                const int h = col >> 6,  d = col & (ASZ - 1);
                ((bf16*)C)[(((long)(b * NHEAD + h)) * ASZ + d) * NSEQ + n] = (bf16)v;
            }
        }
    }
}

// ---------------------------------------------------------------------------
// Flash attention: one wave per 16-row query tile, key blocks of 32,
// logits biased by Gbias[b] (= 0.2*AG + 0.8*AG@AG + VG@VG), online softmax.
// q pre-scaled by QSCALE.  yattn: [B, N, H] bf16.
// ---------------------------------------------------------------------------
__global__ __launch_bounds__(128)
void flash_attn_wmma(const bf16* __restrict__ qb, const bf16* __restrict__ kb,
                     const bf16* __restrict__ vtb, const float* __restrict__ Gb,
                     bf16* __restrict__ yo)
{
    __shared__ bf16 Pl[4][16][32];
    const int tid  = threadIdx.x;
    const int wv   = tid >> 5;
    const int lane = tid & 31;
    const int hl   = lane >> 4;
    const int ln   = lane & 15;
    const int h  = blockIdx.y;
    const int b  = blockIdx.z;
    const int n0 = (blockIdx.x * 4 + wv) * 16;

    const bf16* Q  = qb  + ((long)(b * NHEAD + h)) * NSEQ * ASZ;
    const bf16* Kp = kb  + ((long)(b * NHEAD + h)) * NSEQ * ASZ;
    const bf16* Vt = vtb + ((long)(b * NHEAD + h)) * ASZ * NSEQ;
    const float* G = Gb  + (long)b * NSEQ * NSEQ;

    union { v16bf v; v8bf h2[2]; } aq[2];
#pragma unroll
    for (int t = 0; t < 2; ++t) {
        const bf16* qr = Q + (long)(n0 + ln) * ASZ + t * 32;
        aq[t].h2[0] = *(const v8bf*)(qr + hl * 16);
        aq[t].h2[1] = *(const v8bf*)(qr + hl * 16 + 8);
    }

    float mrow[8], lrow[8];
#pragma unroll
    for (int r = 0; r < 8; ++r) { mrow[r] = -1e30f; lrow[r] = 0.0f; }
    v8f oacc[4] = {};

    for (int j0 = 0; j0 < NSEQ; j0 += 32) {
        v8f s0 = {}, s1 = {};
#pragma unroll
        for (int t = 0; t < 2; ++t) {
            v16bf b0 = *(const v16bf*)(Kp + (long)(j0 + ln) * ASZ + t * 32 + hl * 16);
            s0 = __builtin_amdgcn_wmma_f32_16x16x32_bf16(
                false, aq[t].v, false, b0, (short)0, s0, false, false);
            v16bf b1 = *(const v16bf*)(Kp + (long)(j0 + 16 + ln) * ASZ + t * 32 + hl * 16);
            s1 = __builtin_amdgcn_wmma_f32_16x16x32_bf16(
                false, aq[t].v, false, b1, (short)0, s1, false, false);
        }
#pragma unroll
        for (int r = 0; r < 8; ++r) {
            const int row = n0 + r + hl * 8;
            const float* g = G + (long)row * NSEQ + j0 + ln;
            float p0 = s0[r] + g[0];
            float p1 = s1[r] + g[16];
            float tmax = fmaxf(p0, p1);
#pragma unroll
            for (int off = 1; off < 16; off <<= 1) tmax = fmaxf(tmax, __shfl_xor(tmax, off));
            const float mn = fmaxf(mrow[r], tmax);
            const float e0 = __expf(p0 - mn);
            const float e1 = __expf(p1 - mn);
            float sum = e0 + e1;
#pragma unroll
            for (int off = 1; off < 16; off <<= 1) sum += __shfl_xor(sum, off);
            const float sc = __expf(mrow[r] - mn);
            lrow[r] = lrow[r] * sc + sum;
            mrow[r] = mn;
#pragma unroll
            for (int s = 0; s < 4; ++s) oacc[s][r] *= sc;
            Pl[wv][r + hl * 8][ln]      = (bf16)e0;
            Pl[wv][r + hl * 8][ln + 16] = (bf16)e1;
        }
        __syncthreads();
        union { v16bf v; v8bf h2[2]; } ap;
        ap.h2[0] = *(const v8bf*)(&Pl[wv][ln][hl * 16]);
        ap.h2[1] = *(const v8bf*)(&Pl[wv][ln][hl * 16 + 8]);
#pragma unroll
        for (int s = 0; s < 4; ++s) {
            v16bf bv = *(const v16bf*)(Vt + (long)(s * 16 + ln) * NSEQ + j0 + hl * 16);
            oacc[s] = __builtin_amdgcn_wmma_f32_16x16x32_bf16(
                false, ap.v, false, bv, (short)0, oacc[s], false, false);
        }
        __syncthreads();
    }

#pragma unroll
    for (int r = 0; r < 8; ++r) {
        const float inv = 1.0f / lrow[r];
        const int row = n0 + r + hl * 8;
#pragma unroll
        for (int s = 0; s < 4; ++s)
            yo[((long)(b * NSEQ + row)) * HH + h * ASZ + s * 16 + ln] =
                (bf16)(oacc[s][r] * inv);
    }
}

// ---------------------------------------------------------------------------
// Small support kernels
// ---------------------------------------------------------------------------
__global__ void k_f32_to_bf16(const float* __restrict__ s, bf16* __restrict__ d, int n)
{
    int i = blockIdx.x * blockDim.x + threadIdx.x;
    if (i < n) d[i] = (bf16)s[i];
}

__global__ void k_zero_bf16(bf16* __restrict__ p, long n)
{
    long i = (long)blockIdx.x * blockDim.x + threadIdx.x;
    if (i < n) p[i] = (bf16)0.0f;
}

// var_in construction + bf16 copies of x and var_in
__global__ void k_prep(const float* __restrict__ x, const int* __restrict__ pp,
                       bf16* __restrict__ xb, bf16* __restrict__ vinb)
{
    long i = (long)blockIdx.x * blockDim.x + threadIdx.x;
    if (i >= (long)BB * NSEQ * HH) return;
    const int hc = (int)(i % HH);
    const long bn = i / HH;
    const int n = (int)(bn % NSEQ);
    const int b = (int)(bn / NSEQ);
    xb[i] = (bf16)x[i];
    const bool single = (n == 0) || (n >= 2 && pp[b * NSEQ + n - 2] == 1);
    const int nnext = (n == NSEQ - 1) ? 0 : n + 1;
    const float xn = x[((long)(b * NSEQ + nnext)) * HH + hc];
    float vi;
    if (single) vi = xn;
    else {
        const int nprev = (n == 0) ? NSEQ - 1 : n - 1;
        vi = 0.5f * (x[((long)(b * NSEQ + nprev)) * HH + hc] + xn);
    }
    vinb[i] = (bf16)vi;
}

// edge score: s = lrelu([tvar|tsym] . W_score + b_score), wave per (b,n)
__global__ __launch_bounds__(128)
void k_score(const bf16* __restrict__ tvar, const bf16* __restrict__ tsym,
             const float* __restrict__ Ws, const float* __restrict__ bs,
             float* __restrict__ s)
{
    const int gw = (blockIdx.x * blockDim.x + threadIdx.x) >> 5;
    const int lane = threadIdx.x & 31;
    if (gw >= BB * NSEQ) return;
    const bf16* pv = tvar + (long)gw * HH;
    const bf16* ps = tsym + (long)gw * HH;
    float acc = 0.0f;
    for (int j = lane; j < HH; j += 32)
        acc += (float)pv[j] * Ws[j] + (float)ps[j] * Ws[HH + j];
#pragma unroll
    for (int off = 1; off < 32; off <<= 1) acc += __shfl_xor(acc, off);
    if (lane == 0) {
        const float v = acc + bs[0];
        s[gw] = v > 0.0f ? v : NEG_SLOPE * v;
    }
}

// occurrence-pair score -> scatter into VG (bf16). Disjoint index pairs.
__global__ __launch_bounds__(128)
void k_vg(const float* __restrict__ x, const int* __restrict__ occ,
          const float* __restrict__ Wc, const float* __restrict__ bc,
          bf16* __restrict__ VGb)
{
    const int gw = (blockIdx.x * blockDim.x + threadIdx.x) >> 5;
    const int lane = threadIdx.x & 31;
    if (gw >= BB * MOCC) return;
    const int b = gw / MOCC, m = gw % MOCC;
    const int o0 = occ[(b * MOCC + m) * 2 + 0];
    const int o1 = occ[(b * MOCC + m) * 2 + 1];
    const float* x0 = x + ((long)(b * NSEQ + o0)) * HH;
    const float* x1 = x + ((long)(b * NSEQ + o1)) * HH;
    float acc = 0.0f;
    for (int j = lane; j < HH; j += 32) acc += 0.5f * (x0[j] + x1[j]) * Wc[j];
#pragma unroll
    for (int off = 1; off < 32; off <<= 1) acc += __shfl_xor(acc, off);
    if (lane == 0) {
        float v = acc + bc[0];
        v = v > 0.0f ? v : NEG_SLOPE * v;
        VGb[(long)b * NSEQ * NSEQ + (long)o0 * NSEQ + o1] = (bf16)v;
        VGb[(long)b * NSEQ * NSEQ + (long)o1 * NSEQ + o0] = (bf16)v;
    }
}

// scatter edge scores into dense AG (bf16, zero-initialized, no collisions)
__global__ void k_scatter_ag(const int* __restrict__ pp, const float* __restrict__ s,
                             bf16* __restrict__ AGb)
{
    const int i = blockIdx.x * blockDim.x + threadIdx.x;
    if (i >= BB * NSEQ) return;
    const int b = i >> 10, n = i & (NSEQ - 1);
    if (pp[i] != 1) return;
    const float sv = s[i];
    const bool single = (n == 0) || (n >= 2 && pp[b * NSEQ + n - 2] == 1);
    bf16* A = AGb + (long)b * NSEQ * NSEQ;
    const int im1 = n > 0 ? n - 1 : 0;
    const int ip1 = n < NSEQ - 1 ? n + 1 : NSEQ - 1;
    if (n >= 1) { A[(long)im1 * NSEQ + n] = (bf16)sv; A[(long)n * NSEQ + im1] = (bf16)sv; }
    if (n <= 1 || !single) { A[(long)n * NSEQ + ip1] = (bf16)sv; A[(long)ip1 * NSEQ + n] = (bf16)sv; }
}

// add the 0.2*AG sparse term into the f32 bias matrix (after GEMM passes)
__global__ void k_sparse_add(const int* __restrict__ pp, const float* __restrict__ s,
                             float* __restrict__ Gb)
{
    const int i = blockIdx.x * blockDim.x + threadIdx.x;
    if (i >= BB * NSEQ) return;
    const int b = i >> 10, n = i & (NSEQ - 1);
    if (pp[i] != 1) return;
    const float sv = 0.2f * s[i];
    const bool single = (n == 0) || (n >= 2 && pp[b * NSEQ + n - 2] == 1);
    float* G = Gb + (long)b * NSEQ * NSEQ;
    const int im1 = n > 0 ? n - 1 : 0;
    const int ip1 = n < NSEQ - 1 ? n + 1 : NSEQ - 1;
    if (n >= 1) { G[(long)im1 * NSEQ + n] += sv; G[(long)n * NSEQ + im1] += sv; }
    if (n <= 1 || !single) { G[(long)n * NSEQ + ip1] += sv; G[(long)ip1 * NSEQ + n] += sv; }
}

// per-node path weights + per-batch weight sum (block per batch item)
__global__ void k_weights(const int* __restrict__ pp, float* __restrict__ w,
                          float* __restrict__ wsum)
{
    __shared__ float red[256];
    const int b = blockIdx.x, t = threadIdx.x;
    float local = 0.0f;
    for (int n = t; n < NSEQ; n += 256) {
        float wvv = 0.0f;
        if (pp[b * NSEQ + n] == 1) {
            const bool single = (n == 0) || (n >= 2 && pp[b * NSEQ + n - 2] == 1);
            const bool upper  = (n <= 1) || (!single);
            const bool triple = upper && (n >= 1);
            wvv = triple ? 3.0f : 2.0f;
        }
        w[b * NSEQ + n] = wvv;
        local += wvv;
    }
    red[t] = local;
    __syncthreads();
    for (int st = 128; st > 0; st >>= 1) {
        if (t < st) red[t] += red[t + st];
        __syncthreads();
    }
    if (t == 0) wsum[b] = red[0];
}

// cbar[b,j] = sum_n w * concat(var_y, y)[b,n,j] / wsum[b]
__global__ void k_cbar(const float* __restrict__ y, const int* __restrict__ pp,
                       const float* __restrict__ w, const float* __restrict__ wsum,
                       float* __restrict__ cbar)
{
    const int j = blockIdx.x * blockDim.x + threadIdx.x;
    const int b = blockIdx.y;
    if (j >= 2 * HH) return;
    float acc = 0.0f;
    for (int n = 0; n < NSEQ; ++n) {
        const float wv = w[b * NSEQ + n];
        if (wv == 0.0f) continue;
        float val;
        if (j < HH) {
            const bool single = (n == 0) || (n >= 2 && pp[b * NSEQ + n - 2] == 1);
            const int nx = (n == NSEQ - 1) ? 0 : n + 1;
            const float yn = y[((long)(b * NSEQ + nx)) * HH + j];
            if (single) val = yn;
            else {
                const int npv = (n == 0) ? NSEQ - 1 : n - 1;
                val = 0.5f * (y[((long)(b * NSEQ + npv)) * HH + j] + yn);
            }
        } else {
            val = y[((long)(b * NSEQ + n)) * HH + (j - HH)];
        }
        acc += wv * val;
    }
    cbar[b * 2 * HH + j] = acc / wsum[b];
}

// pm[b,o] = cbar[b,:] . W_atom[o,:] + b_atom[o]   (wave per output)
__global__ __launch_bounds__(128)
void k_pm(const float* __restrict__ cbar, const float* __restrict__ Wa,
          const float* __restrict__ ba, float* __restrict__ pm)
{
    const int gw = (blockIdx.x * blockDim.x + threadIdx.x) >> 5;
    const int lane = threadIdx.x & 31;
    if (gw >= BB * HH) return;
    const int b = gw / HH, o = gw % HH;
    const float* c  = cbar + b * 2 * HH;
    const float* wr = Wa + (long)o * 2 * HH;
    float acc = 0.0f;
    for (int j = lane; j < 2 * HH; j += 32) acc += c[j] * wr[j];
#pragma unroll
    for (int off = 1; off < 32; off <<= 1) acc += __shfl_xor(acc, off);
    if (lane == 0) pm[gw] = acc + ba[o];
}

__global__ void k_bcast(const float* __restrict__ pm, float* __restrict__ out)
{
    long i = (long)blockIdx.x * blockDim.x + threadIdx.x;
    if (i >= (long)BB * NSEQ * HH) return;
    const int hc = (int)(i % HH);
    const int b  = (int)(i / ((long)NSEQ * HH));
    out[i] = pm[b * HH + hc];
}

// ---------------------------------------------------------------------------
extern "C" void kernel_launch(void* const* d_in, const int* in_sizes, int n_in,
                              void* d_out, int out_size, void* d_ws, size_t ws_size,
                              hipStream_t stream)
{
    (void)in_sizes; (void)n_in; (void)out_size; (void)ws_size;
    const float* x       = (const float*)d_in[0];
    const float* W_var   = (const float*)d_in[3];  const float* b_var  = (const float*)d_in[4];
    const float* W_sym   = (const float*)d_in[5];  const float* b_sym  = (const float*)d_in[6];
    const float* W_score = (const float*)d_in[7];  const float* b_sc   = (const float*)d_in[8];
    const float* W_cross = (const float*)d_in[9];  const float* b_cr   = (const float*)d_in[10];
    const float* W_atom  = (const float*)d_in[11]; const float* b_atom = (const float*)d_in[12];
    const float* W_q     = (const float*)d_in[13]; const float* b_q    = (const float*)d_in[14];
    const float* W_k     = (const float*)d_in[15]; const float* b_k    = (const float*)d_in[16];
    const float* W_v     = (const float*)d_in[17]; const float* b_v    = (const float*)d_in[18];
    const float* W_o     = (const float*)d_in[19]; const float* b_o    = (const float*)d_in[20];
    const int*   pp      = (const int*)d_in[21];
    const int*   occ     = (const int*)d_in[24];
    float* out = (float*)d_out;

    const long BNH = (long)BB * NSEQ * HH;
    const long BNN = (long)BB * NSEQ * NSEQ;

    char* cur = (char*)d_ws;
    auto alloc = [&](size_t bytes) -> char* {
        char* p = cur;
        cur += (bytes + 255) & ~(size_t)255;
        return p;
    };
    bf16* xb     = (bf16*)alloc(BNH * 2);
    bf16* vinb   = (bf16*)alloc(BNH * 2);
    bf16* Wb     = (bf16*)alloc((size_t)6 * HH * HH * 2);
    bf16* Wb_var = Wb + 0L * HH * HH;
    bf16* Wb_sym = Wb + 1L * HH * HH;
    bf16* Wb_q   = Wb + 2L * HH * HH;
    bf16* Wb_k   = Wb + 3L * HH * HH;
    bf16* Wb_v   = Wb + 4L * HH * HH;
    bf16* Wb_o   = Wb + 5L * HH * HH;
    bf16* tvar   = (bf16*)alloc(BNH * 2);
    bf16* tsym   = (bf16*)alloc(BNH * 2);
    float* sarr  = (float*)alloc((size_t)BB * NSEQ * 4);
    bf16* AGb    = (bf16*)alloc(BNN * 2);
    bf16* VGb    = (bf16*)alloc(BNN * 2);
    float* Gbias = (float*)alloc(BNN * 4);
    bf16* qb     = (bf16*)alloc(BNH * 2);
    bf16* kb     = (bf16*)alloc(BNH * 2);
    bf16* vtb    = (bf16*)alloc(BNH * 2);
    bf16* yattn  = (bf16*)alloc(BNH * 2);
    float* y     = (float*)alloc(BNH * 4);
    float* warr  = (float*)alloc((size_t)BB * NSEQ * 4);
    float* wsum  = (float*)alloc((size_t)BB * 4);
    float* cbar  = (float*)alloc((size_t)BB * 2 * HH * 4);
    float* pmv   = (float*)alloc((size_t)BB * HH * 4);

    const int WN = HH * HH;
    const int cvb = (WN + 255) / 256;
    k_f32_to_bf16<<<cvb, 256, 0, stream>>>(W_var, Wb_var, WN);
    k_f32_to_bf16<<<cvb, 256, 0, stream>>>(W_sym, Wb_sym, WN);
    k_f32_to_bf16<<<cvb, 256, 0, stream>>>(W_q,   Wb_q,   WN);
    k_f32_to_bf16<<<cvb, 256, 0, stream>>>(W_k,   Wb_k,   WN);
    k_f32_to_bf16<<<cvb, 256, 0, stream>>>(W_v,   Wb_v,   WN);
    k_f32_to_bf16<<<cvb, 256, 0, stream>>>(W_o,   Wb_o,   WN);

    k_prep<<<(int)((BNH + 255) / 256), 256, 0, stream>>>(x, pp, xb, vinb);

    // var/sym features -> tanh -> edge scores
    dim3 gB(HH / 64, (BB * NSEQ) / 64, 1);
    gemm_bf16_wmma<ACT_TANH, SM_BF16, 0><<<gB, 128, 0, stream>>>(
        vinb, HH, 0, Wb_var, HH, 0, tvar, HH, 0, b_var, 1.0f, HH);
    gemm_bf16_wmma<ACT_TANH, SM_BF16, 0><<<gB, 128, 0, stream>>>(
        xb, HH, 0, Wb_sym, HH, 0, tsym, HH, 0, b_sym, 1.0f, HH);
    k_score<<<(BB * NSEQ) / 4, 128, 0, stream>>>(tvar, tsym, W_score, b_sc, sarr);

    // graphs: zero, scatter, square, combine
    k_zero_bf16<<<(int)((BNN + 255) / 256), 256, 0, stream>>>(AGb, BNN);
    k_zero_bf16<<<(int)((BNN + 255) / 256), 256, 0, stream>>>(VGb, BNN);
    k_scatter_ag<<<(BB * NSEQ) / 256, 256, 0, stream>>>(pp, sarr, AGb);
    k_vg<<<(BB * MOCC) / 4, 128, 0, stream>>>(x, occ, W_cross, b_cr, VGb);

    dim3 gG(NSEQ / 64, NSEQ / 64, BB);
    const long sNN = (long)NSEQ * NSEQ;
    gemm_bf16_wmma<ACT_NONE, SM_F32, 0><<<gG, 128, 0, stream>>>(
        AGb, NSEQ, sNN, AGb, NSEQ, sNN, Gbias, NSEQ, sNN, nullptr, 0.8f, NSEQ);
    gemm_bf16_wmma<ACT_NONE, SM_F32, 1><<<gG, 128, 0, stream>>>(
        VGb, NSEQ, sNN, VGb, NSEQ, sNN, Gbias, NSEQ, sNN, nullptr, 1.0f, NSEQ);
    k_sparse_add<<<(BB * NSEQ) / 256, 256, 0, stream>>>(pp, sarr, Gbias);

    // q/k/v projections (q pre-scaled; v stored head-transposed)
    gemm_bf16_wmma<ACT_NONE, SM_HEADS, 0><<<gB, 128, 0, stream>>>(
        xb, HH, 0, Wb_q, HH, 0, qb, 0, 0, b_q, QSCALE, HH);
    gemm_bf16_wmma<ACT_NONE, SM_HEADS, 0><<<gB, 128, 0, stream>>>(
        xb, HH, 0, Wb_k, HH, 0, kb, 0, 0, b_k, 1.0f, HH);
    gemm_bf16_wmma<ACT_NONE, SM_HEADS_T, 0><<<gB, 128, 0, stream>>>(
        xb, HH, 0, Wb_v, HH, 0, vtb, 0, 0, b_v, 1.0f, HH);

    dim3 gA(NSEQ / 64, NHEAD, BB);
    flash_attn_wmma<<<gA, 128, 0, stream>>>(qb, kb, vtb, Gbias, yattn);

    // output projection (f32 result, needed for rolls)
    gemm_bf16_wmma<ACT_NONE, SM_F32, 0><<<gB, 128, 0, stream>>>(
        yattn, HH, 0, Wb_o, HH, 0, y, HH, 0, b_o, 1.0f, HH);

    // path embedding: linear -> reduce first, then one small matvec
    k_weights<<<BB, 256, 0, stream>>>(pp, warr, wsum);
    dim3 gC((2 * HH + 127) / 128, BB);
    k_cbar<<<gC, 128, 0, stream>>>(y, pp, warr, wsum, cbar);
    k_pm<<<(BB * HH) / 4, 128, 0, stream>>>(cbar, W_atom, b_atom, pmv);
    k_bcast<<<(int)((BNH + 255) / 256), 256, 0, stream>>>(pmv, out);
}
